// Net_20349555048543
// MI455X (gfx1250) — matrix operand order repaired and checked by reading, not compiled, not run on hardware
//
#include <hip/hip_runtime.h>
#include <hip/hip_bf16.h>

#define DEVINL __device__ __forceinline__

typedef __bf16 bf16_t;
typedef __attribute__((ext_vector_type(16))) __bf16 bf16x16;
typedef __attribute__((ext_vector_type(8)))  __bf16 bf16x8;
typedef __attribute__((ext_vector_type(8)))  float  f32x8;
typedef __attribute__((ext_vector_type(4)))  unsigned int u32x4;
typedef __attribute__((ext_vector_type(8)))  unsigned int u32x8;

static constexpr int B_ = 4, C0 = 64, C4 = 256, H_ = 96, W_ = 256;
static constexpr int HW = H_ * W_;                       // 24576
static constexpr int BH = B_ * H_;                       // 384
static constexpr long long X_ELEMS = (long long)B_ * C0 * HW;   // 6291456
static constexpr long long Y_ELEMS = (long long)B_ * C4 * HW;   // 25165824
static constexpr long long M_ELEMS = (long long)BH * W_ * W_;   // 25165824

// ---------------- WMMA helpers (CDNA5 v_wmma_f32_16x16x32_bf16) ---------------
DEVINL f32x8 fzero() {
  f32x8 v;
  #pragma unroll
  for (int i = 0; i < 8; ++i) v[i] = 0.f;
  return v;
}
DEVINL bf16x16 hzero() {
  bf16x16 v;
  #pragma unroll
  for (int i = 0; i < 16; ++i) v[i] = (bf16_t)0.f;
  return v;
}

DEVINL f32x8 wmma_bf16(bf16x16 a, bf16x16 b, f32x8 c) {
  // D = A(16x32) x B(32x16) + C ; B fed as (N,K) row-major fragments
  return __builtin_amdgcn_wmma_f32_16x16x32_bf16(false, a, false, b, (short)0, c,
                                                 false, false);
}

// A fragment: lane L -> row = L&15; 16 contiguous K at hf*16
DEVINL bf16x16 make_afrag(const bf16_t* p) {
  bf16x8 lo = *(const bf16x8*)p;
  bf16x8 hi = *(const bf16x8*)(p + 8);
  bf16x16 v;
  #pragma unroll
  for (int i = 0; i < 8; ++i) { v[i] = lo[i]; v[i + 8] = hi[i]; }
  return v;
}
DEVINL bf16x16 load_a(const bf16_t* tile, int pitch, int lane) {
  const int r = lane & 15, hf = lane >> 4;
  return make_afrag(tile + (size_t)r * pitch + hf * 16);
}
// B fragment: lane L -> col n = L&15; 16 contiguous K at hf*16 ((N,K) row-major)
DEVINL bf16x16 load_b(const bf16_t* tile, int pitch, int lane) {
  const int n = lane & 15, hf = lane >> 4;
  return *(const bf16x16*)(tile + (size_t)n * pitch + hf * 16);
}

// -------- TDM: async 2D tile (256x256 bf16) global -> LDS, pitch 258 ---------
// D# group0/group1 per CDNA5 ISA; pad 1 DWORD every 128
// DWORDs turns the 512B rows into 516B (= 258 bf16) rows in LDS.
DEVINL void tdm_load_m_slice(const void* gptr, unsigned lds_off) {
  const unsigned long long ga = (unsigned long long)(uintptr_t)gptr;
  u32x4 g0;
  g0[0] = 1u;                                   // count=1, user descriptor
  g0[1] = lds_off;                              // LDS byte address
  g0[2] = (unsigned)ga;                         // global_addr[31:0]
  g0[3] = ((unsigned)(ga >> 32) & 0x01FFFFFFu)  // global_addr[56:32]
          | (2u << 30);                         // type = 2 ("image")
  u32x8 g1;
  g1[0] = (1u << 16)      // data_size = 1 -> 2 bytes
        | (1u << 20)      // pad_enable
        | (6u << 22)      // pad_interval: 128 DWORDs
        | (0u << 25);     // pad_amount: 1 DWORD
  g1[1] = (256u << 16);   // tensor_dim0[15:0] = 256 (bits 63:48)
  g1[2] = (256u << 16);   // tensor_dim0 hi = 0 | tensor_dim1[15:0] = 256
  g1[3] = (256u << 16);   // tensor_dim1 hi = 0 | tile_dim0 = 256
  g1[4] = 256u;           // tile_dim1 = 256, tile_dim2 = 0
  g1[5] = 256u;           // tensor_dim0_stride = 256 elements
  g1[6] = 0u;
  g1[7] = 0u;
  asm volatile("tensor_load_to_lds %0, %1" :: "s"(g0), "s"(g1) : "memory");
}

// ------------------------------- BN stats ------------------------------------
__global__ void k_bn_stats(const float* __restrict__ catL, const float* __restrict__ catR,
                           float* __restrict__ meanArr, float* __restrict__ rstdArr) {
  const int c = blockIdx.x, side = blockIdx.y;
  const float* in = side ? catR : catL;
  float s = 0.f, sq = 0.f;
  for (int bb = 0; bb < B_; ++bb) {
    const float* p = in + (size_t)(bb * C4 + c) * HW;
    for (int i = threadIdx.x; i < HW; i += 256) { float v = p[i]; s += v; sq += v * v; }
  }
  __shared__ float rs[256], rq[256];
  rs[threadIdx.x] = s; rq[threadIdx.x] = sq;
  __syncthreads();
  for (int off = 128; off > 0; off >>= 1) {
    if (threadIdx.x < off) { rs[threadIdx.x] += rs[threadIdx.x + off];
                             rq[threadIdx.x] += rq[threadIdx.x + off]; }
    __syncthreads();
  }
  if (threadIdx.x == 0) {
    const float n = (float)B_ * (float)HW;
    float m   = rs[0] / n;
    float var = rq[0] / n - m * m;
    meanArr[side * C4 + c] = m;
    rstdArr[side * C4 + c] = rsqrtf(var + 1e-5f);
  }
}

// --------------- BN apply: write y (f32, NCHW) + y_bf (bf16, (b,g,h,w,ic)) ----
__global__ void k_bn_apply(const float* __restrict__ cat, const float* __restrict__ gamma,
                           const float* __restrict__ beta, const float* __restrict__ meanArr,
                           const float* __restrict__ rstdArr,
                           float* __restrict__ yf, bf16_t* __restrict__ ybf) {
  const int wblk = blockIdx.x, h = blockIdx.y, bg = blockIdx.z;
  const int b = bg >> 2, g = bg & 3;
  __shared__ float tile[64 * 65];
  for (int l = threadIdx.x; l < 4096; l += 256) {
    const int ic = l >> 6, wl = l & 63;
    const int c = g * 64 + ic;
    const size_t idx = ((size_t)(b * C4 + c) * H_ + h) * W_ + wblk * 64 + wl;
    float v = (cat[idx] - meanArr[c]) * rstdArr[c] * gamma[c] + beta[c];
    yf[idx] = v;
    tile[ic * 65 + wl] = v;
  }
  __syncthreads();
  bf16_t* dst = ybf + ((size_t)(bg * H_ + h) * W_ + wblk * 64) * 64;
  for (int l = threadIdx.x; l < 4096; l += 256) {
    const int wl = l >> 6, ic = l & 63;
    dst[(size_t)wl * 64 + ic] = (bf16_t)tile[ic * 65 + wl];
  }
}

// --------- weight repack: OIHW f32 -> (oc, tap, ic) bf16 for both convs -------
__global__ void k_conv_w(const float* __restrict__ w1, const float* __restrict__ w2,
                         bf16_t* __restrict__ w1b, bf16_t* __restrict__ w2b) {
  const int idx = blockIdx.x * 256 + threadIdx.x;
  if (idx >= C4 * 576) return;
  const int o = idx / 576, r = idx % 576, tap = r / 64, ic = r % 64;
  const size_t src = ((size_t)o * 64 + ic) * 9 + tap;
  w1b[idx] = (bf16_t)w1[src];
  w2b[idx] = (bf16_t)w2[src];
}

__global__ void k_conv_x(const float* __restrict__ xl, const float* __restrict__ xr,
                         bf16_t* __restrict__ xlb, bf16_t* __restrict__ xrb, long long n) {
  const long long i = (long long)blockIdx.x * 256 + threadIdx.x;
  if (i < n) { xlb[i] = (bf16_t)xl[i]; xrb[i] = (bf16_t)xr[i]; }
}

// --------------- grouped 3x3 conv as 9 shifted WMMA GEMMs --------------------
// mode 0: r = leaky(conv1(y_bf)+b1) -> dstbf (b,g,h,w,oc)
// mode 1: y += conv2(r_bf)+b2       -> yio (NCHW f32, in-place)
__global__ void k_conv3x3(const bf16_t* __restrict__ src, const bf16_t* __restrict__ wbf,
                          const float* __restrict__ bias,
                          bf16_t* __restrict__ dstbf, float* __restrict__ yio, int mode) {
  const int wt = blockIdx.x, h = blockIdx.y, bg = blockIdx.z;
  const int b = bg >> 2, g = bg & 3;
  const int lane = threadIdx.x & 31, wid = threadIdx.x >> 5;
  const int hf = lane >> 4, ln = lane & 15;
  const int mi = wid & 3;              // oc tile within group (16 oc)
  const int nj0 = (wid >> 2) * 4;      // 4 n-tiles of 16 w
  const int w0 = wt * 128;
  const int ocb = mi * 16;
  const bf16_t* wrow = wbf + (size_t)(g * 64 + ocb) * 576;

  f32x8 acc[4];
  #pragma unroll
  for (int j = 0; j < 4; ++j) acc[j] = fzero();

  for (int kh = 0; kh < 3; ++kh) {
    const int hh = h + kh - 1;
    const bool rowok = (hh >= 0) && (hh < H_);
    const bf16_t* srow = src + (size_t)(bg * H_ + (rowok ? hh : 0)) * W_ * 64;
    for (int kw = 0; kw < 3; ++kw) {
      const int tap = kh * 3 + kw;
      const int dw = kw - 1;
      #pragma unroll
      for (int ks = 0; ks < 2; ++ks) {
        const int ic0 = ks * 32;
        const bf16x16 afrag = make_afrag(wrow + (size_t)ln * 576 + tap * 64 + ic0 + hf * 16);
        #pragma unroll
        for (int j = 0; j < 4; ++j) {
          const int w = w0 + (nj0 + j) * 16 + ln + dw;
          bf16x16 bfrag = hzero();
          if (rowok && (unsigned)w < (unsigned)W_)
            bfrag = *(const bf16x16*)(srow + (size_t)w * 64 + ic0 + hf * 16);
          acc[j] = wmma_bf16(afrag, bfrag, acc[j]);
        }
      }
    }
  }
  const int oc_lane = ocb + hf * 8;    // 8 contiguous oc per lane
  if (mode == 0) {
    #pragma unroll
    for (int j = 0; j < 4; ++j) {
      const int w = w0 + (nj0 + j) * 16 + ln;
      bf16x8 outv;
      #pragma unroll
      for (int r = 0; r < 8; ++r) {
        float v = acc[j][r] + bias[g * 64 + oc_lane + r];
        v = (v >= 0.f) ? v : 0.1f * v;
        outv[r] = (bf16_t)v;
      }
      *(bf16x8*)(dstbf + ((size_t)(bg * H_ + h) * W_ + w) * 64 + oc_lane) = outv;
    }
  } else {
    #pragma unroll
    for (int j = 0; j < 4; ++j) {
      const int w = w0 + (nj0 + j) * 16 + ln;
      #pragma unroll
      for (int r = 0; r < 8; ++r) {
        const int c = g * 64 + oc_lane + r;
        const size_t idx = ((size_t)(b * C4 + c) * H_ + h) * W_ + w;
        yio[idx] = yio[idx] + acc[j][r] + bias[c];
      }
    }
  }
}

// -------- grouped 1x1 conv (memory-bound; VALU) -> Qtmp (b,h,c,w) f32 --------
__global__ void k_conv1x1(const float* __restrict__ yf, const float* __restrict__ w,
                          const float* __restrict__ bias, float* __restrict__ qtmp) {
  const int bh = blockIdx.x, b = bh / H_, h = bh % H_;
  __shared__ float wl[64 * 64];
  __shared__ float bl[64];
  for (int i = threadIdx.x; i < 4096; i += 256) wl[i] = w[i];
  if (threadIdx.x < 64) bl[threadIdx.x] = bias[threadIdx.x];
  __syncthreads();
  const int t = threadIdx.x;  // w position
  for (int g = 0; g < 4; ++g) {
    float acc[16];
    #pragma unroll
    for (int o = 0; o < 16; ++o) acc[o] = bl[g * 16 + o];
    for (int ic = 0; ic < 64; ++ic) {
      const float yv = yf[((size_t)(b * C4 + g * 64 + ic) * H_ + h) * W_ + t];
      #pragma unroll
      for (int o = 0; o < 16; ++o) acc[o] += wl[(g * 16 + o) * 64 + ic] * yv;
    }
    #pragma unroll
    for (int o = 0; o < 16; ++o)
      qtmp[((size_t)bh * 64 + g * 16 + o) * W_ + t] = acc[o];
  }
}

// -------- subtract mean over w; transpose to (b,h,w,c) bf16 via LDS ----------
__global__ void k_rowmean(const float* __restrict__ qtmp, bf16_t* __restrict__ qbf) {
  extern __shared__ char dsm[];
  float* tile = (float*)dsm;               // 64 x 257 f32
  __shared__ float mn[64];
  const int bh = blockIdx.x;
  const float* src = qtmp + (size_t)bh * 64 * W_;
  for (int l = threadIdx.x; l < 64 * W_; l += 256) {
    const int c = l >> 8, w = l & 255;
    tile[c * 257 + w] = src[l];
  }
  __syncthreads();
  if (threadIdx.x < 64) {
    float s = 0.f;
    for (int w = 0; w < W_; ++w) s += tile[threadIdx.x * 257 + w];
    mn[threadIdx.x] = s * (1.f / (float)W_);
  }
  __syncthreads();
  bf16_t* dst = qbf + (size_t)bh * W_ * 64;
  for (int l = threadIdx.x; l < 64 * W_; l += 256) {
    const int w = l >> 6, c = l & 63;
    dst[l] = (bf16_t)(tile[c * 257 + w] - mn[c]);   // l == w*64 + c
  }
}

// ------ S = Q^T K via WMMA (S in 257-pitch LDS), dual softmax -> M1, M2 ------
__global__ void k_score_softmax(const bf16_t* __restrict__ qbf, const bf16_t* __restrict__ kbf,
                                bf16_t* __restrict__ m1, bf16_t* __restrict__ m2) {
  extern __shared__ char dsm[];
  float* S = (float*)dsm;                  // 256 x 257 f32
  __shared__ float rmax[256], rinv[256], cmax[256], cinv[256];
  const int bh = blockIdx.x;
  const bf16_t* Q = qbf + (size_t)bh * W_ * 64;
  const bf16_t* K = kbf + (size_t)bh * W_ * 64;
  const int lane = threadIdx.x & 31, wid = threadIdx.x >> 5;
  const int ln = lane & 15;

  bf16x16 afr[2][2];
  #pragma unroll
  for (int mm = 0; mm < 2; ++mm) {
    #pragma unroll
    for (int ks = 0; ks < 2; ++ks) {
      afr[mm][ks] = load_a(Q + (size_t)(wid * 32 + mm * 16) * 64 + ks * 32, 64, lane);
    }
  }

  for (int ni = 0; ni < 16; ++ni) {
    f32x8 acc0 = fzero(), acc1 = fzero();
    #pragma unroll
    for (int ks = 0; ks < 2; ++ks) {
      const bf16x16 bfr = load_b(K + (size_t)(ni * 16) * 64 + ks * 32, 64, lane);
      acc0 = wmma_bf16(afr[0][ks], bfr, acc0);
      acc1 = wmma_bf16(afr[1][ks], bfr, acc1);
    }
    const int col = ni * 16 + ln;
    const int rbase = wid * 32 + (lane >> 4) * 8;
    #pragma unroll
    for (int r = 0; r < 8; ++r) {
      S[(rbase + r) * 257 + col]      = acc0[r];
      S[(rbase + 16 + r) * 257 + col] = acc1[r];
    }
  }
  __syncthreads();
  {
    const int t = threadIdx.x;
    float mx = -3.4e38f;
    for (int k2 = 0; k2 < W_; ++k2) mx = fmaxf(mx, S[t * 257 + k2]);
    float s = 0.f;
    for (int k2 = 0; k2 < W_; ++k2) s += __expf(S[t * 257 + k2] - mx);
    rmax[t] = mx; rinv[t] = 1.f / s;
    float cx = -3.4e38f;
    for (int q2 = 0; q2 < W_; ++q2) cx = fmaxf(cx, S[q2 * 257 + t]);
    float cs = 0.f;
    for (int q2 = 0; q2 < W_; ++q2) cs += __expf(S[q2 * 257 + t] - cx);
    cmax[t] = cx; cinv[t] = 1.f / cs;
  }
  __syncthreads();
  bf16_t* M1 = m1 + (size_t)bh * 65536;
  bf16_t* M2 = m2 + (size_t)bh * 65536;
  for (int l = threadIdx.x; l < 65536; l += 256) {
    const int q = l >> 8, k2 = l & 255;
    M1[l] = (bf16_t)(__expf(S[q * 257 + k2] - rmax[q]) * rinv[q]);
  }
  for (int l = threadIdx.x; l < 65536; l += 256) {
    const int k2 = l >> 8, q = l & 255;
    M2[l] = (bf16_t)(__expf(S[q * 257 + k2] - cmax[k2]) * cinv[k2]);
  }
}

// ------------- band-relaxed validity maps: V_left/V_right -> tanh ------------
// M1/M2 slices staged to LDS by the Tensor Data Mover (async, TENSORcnt),
// with TDM row padding producing a conflict-free 258-halfword pitch.
__global__ void k_vmaps(const bf16_t* __restrict__ m1, const bf16_t* __restrict__ m2,
                        float* __restrict__ vl, float* __restrict__ vr) {
  extern __shared__ char dsm[];
  bf16_t* m1s = (bf16_t*)dsm;            // natural [q][k], pitch 258
  bf16_t* m2s = m1s + 256 * 258;         // natural [k][q], pitch 258
  const int bh = blockIdx.x;
  const bf16_t* M1 = m1 + (size_t)bh * 65536;
  const bf16_t* M2 = m2 + (size_t)bh * 65536;

  if (threadIdx.x < 32) {                // wave 0 issues both DMAs
    tdm_load_m_slice((const void*)M1, (unsigned)(uintptr_t)m1s);
    tdm_load_m_slice((const void*)M2, (unsigned)(uintptr_t)m2s);
    __builtin_amdgcn_s_wait_tensorcnt(0);
  }
  __syncthreads();

  const int t = threadIdx.x;
  const int dlo = (t >= 2) ? -2 : -t;
  const int dhi = (t <= W_ - 3) ? 2 : (W_ - 1 - t);
  float accl = 0.f;
  for (int k = 0; k < W_; ++k) {
    float ws = 0.f;
    for (int d = dlo; d <= dhi; ++d) ws += (float)m1s[(t + d) * 258 + k];
    accl += ws * (float)m2s[k * 258 + t];
  }
  vl[(size_t)bh * W_ + t] = tanhf(5.f * accl);
  float accr = 0.f;
  for (int q = 0; q < W_; ++q) {
    float ws = 0.f;
    for (int d = dlo; d <= dhi; ++d) ws += (float)m2s[(t + d) * 258 + q];
    accr += ws * (float)m1s[q * 258 + t];
  }
  vr[(size_t)bh * W_ + t] = tanhf(5.f * accr);
}

// ------- out = x*(1-V) + (x_other @ M^T)*V  (WMMA, fused blend epilogue) -----
__global__ void k_xt_blend(const bf16_t* __restrict__ xlb, const bf16_t* __restrict__ xrb,
                           const bf16_t* __restrict__ m1, const bf16_t* __restrict__ m2,
                           const float* __restrict__ vl, const float* __restrict__ vr,
                           const float* __restrict__ xl, const float* __restrict__ xr,
                           float* __restrict__ out) {
  const int bh = blockIdx.x, side = blockIdx.y;
  const int b = bh / H_, h = bh % H_;
  const bf16_t* A  = (side == 0 ? xrb : xlb) + (size_t)b * C0 * HW + (size_t)h * W_;
  const bf16_t* Bm = (side == 0 ? m1 : m2) + (size_t)bh * 65536;
  const float*  V  = (side == 0 ? vl : vr) + (size_t)bh * W_;
  const float*  X  = (side == 0 ? xl : xr);
  float* O = out + (side == 0 ? 0 : (size_t)X_ELEMS) + (size_t)b * C0 * HW + (size_t)h * W_;
  const int lane = threadIdx.x & 31, wid = threadIdx.x >> 5;
  const int hf = lane >> 4, ln = lane & 15;

  f32x8 acc[4][2];
  #pragma unroll
  for (int mi = 0; mi < 4; ++mi) { acc[mi][0] = fzero(); acc[mi][1] = fzero(); }

  for (int ks = 0; ks < 8; ++ks) {
    if (ks + 1 < 8)  // pull next K-step of the M operand toward the caches
      __builtin_prefetch(Bm + (size_t)(wid * 2 * 16 + ln) * 256 + (ks + 1) * 32, 0, 0);
    bf16x16 bfr[2];
    #pragma unroll
    for (int j = 0; j < 2; ++j)
      bfr[j] = load_b(Bm + (size_t)((wid * 2 + j) * 16) * 256 + ks * 32, 256, lane);
    #pragma unroll
    for (int mi = 0; mi < 4; ++mi) {
      const bf16x16 afr = load_a(A + (size_t)(mi * 16) * HW + ks * 32, HW, lane);
      acc[mi][0] = wmma_bf16(afr, bfr[0], acc[mi][0]);
      acc[mi][1] = wmma_bf16(afr, bfr[1], acc[mi][1]);
    }
  }
  #pragma unroll
  for (int j = 0; j < 2; ++j) {
    const int n = (wid * 2 + j) * 16 + ln;   // output column (q or k)
    const float v = V[n];
    #pragma unroll
    for (int mi = 0; mi < 4; ++mi) {
      #pragma unroll
      for (int r = 0; r < 8; ++r) {
        const int c = mi * 16 + hf * 8 + r;
        const size_t idx = (size_t)c * HW + n;
        O[idx] = X[(size_t)(b * C0 + c) * HW + (size_t)h * W_ + n] * (1.f - v)
               + acc[mi][j][r] * v;
      }
    }
  }
}

// =============================== host launcher ===============================
extern "C" void kernel_launch(void* const* d_in, const int* in_sizes, int n_in,
                              void* d_out, int out_size, void* d_ws, size_t ws_size,
                              hipStream_t stream) {
  (void)in_sizes; (void)n_in; (void)out_size; (void)ws_size;
  const float* x_left  = (const float*)d_in[0];
  const float* x_right = (const float*)d_in[1];
  const float* catL    = (const float*)d_in[2];
  const float* catR    = (const float*)d_in[3];
  const float* gamma   = (const float*)d_in[4];
  const float* beta    = (const float*)d_in[5];
  const float* rb_w1   = (const float*)d_in[6];
  const float* rb_b1   = (const float*)d_in[7];
  const float* rb_w2   = (const float*)d_in[8];
  const float* rb_b2   = (const float*)d_in[9];
  const float* bq_w    = (const float*)d_in[10];
  const float* bq_b    = (const float*)d_in[11];
  const float* bs_w    = (const float*)d_in[12];
  const float* bs_b    = (const float*)d_in[13];
  float* out = (float*)d_out;

  char* ws = (char*)d_ws;
  size_t off = 0;
  auto alloc = [&](size_t bytes) -> char* {
    char* p = ws + off;
    off += (bytes + 255) & ~(size_t)255;
    return p;
  };
  float*  meanArr = (float*)alloc(2 * C4 * 4);
  float*  rstdArr = (float*)alloc(2 * C4 * 4);
  float*  VL      = (float*)alloc((size_t)BH * W_ * 4);
  float*  VR      = (float*)alloc((size_t)BH * W_ * 4);
  bf16_t* w1b     = (bf16_t*)alloc((size_t)C4 * 576 * 2);
  bf16_t* w2b     = (bf16_t*)alloc((size_t)C4 * 576 * 2);
  bf16_t* xlb     = (bf16_t*)alloc((size_t)X_ELEMS * 2);
  bf16_t* xrb     = (bf16_t*)alloc((size_t)X_ELEMS * 2);
  bf16_t* Qbf     = (bf16_t*)alloc((size_t)X_ELEMS * 2);
  bf16_t* Kbf     = (bf16_t*)alloc((size_t)X_ELEMS * 2);
  bf16_t* ybf     = (bf16_t*)alloc((size_t)Y_ELEMS * 2);
  // region A: y f32 during conv phase, then reused for M1/M2 (bf16) afterwards
  char*   regA    = alloc((size_t)Y_ELEMS * 4);
  float*  yf      = (float*)regA;
  bf16_t* M1      = (bf16_t*)regA;
  bf16_t* M2      = (bf16_t*)(regA + (size_t)M_ELEMS * 2);
  // region B: r bf16 during convs, reused as Qtmp f32 for the 1x1 stage
  char*   regB    = alloc((size_t)Y_ELEMS * 2);
  bf16_t* rbf     = (bf16_t*)regB;
  float*  qtmp    = (float*)regB;

  // ---- precompute (independent) ----
  k_bn_stats<<<dim3(C4, 2), 256, 0, stream>>>(catL, catR, meanArr, rstdArr);
  k_conv_w<<<dim3((C4 * 576 + 255) / 256), 256, 0, stream>>>(rb_w1, rb_w2, w1b, w2b);
  k_conv_x<<<dim3((unsigned)((X_ELEMS + 255) / 256)), 256, 0, stream>>>(
      x_left, x_right, xlb, xrb, X_ELEMS);

  // ---- two branches (sequential; buffers reused) ----
  for (int side = 0; side < 2; ++side) {
    const float* cat   = side ? catR : catL;
    const float* w1x1  = side ? bs_w : bq_w;
    const float* b1x1  = side ? bs_b : bq_b;
    bf16_t*      qkdst = side ? Kbf : Qbf;
    k_bn_apply<<<dim3(4, H_, 16), 256, 0, stream>>>(
        cat, gamma, beta, meanArr + side * C4, rstdArr + side * C4, yf, ybf);
    k_conv3x3<<<dim3(2, H_, 16), 256, 0, stream>>>(ybf, w1b, rb_b1, rbf, nullptr, 0);
    k_conv3x3<<<dim3(2, H_, 16), 256, 0, stream>>>(rbf, w2b, rb_b2, nullptr, yf, 1);
    k_conv1x1<<<dim3(BH), 256, 0, stream>>>(yf, w1x1, b1x1, qtmp);
    k_rowmean<<<dim3(BH), 256, 64 * 257 * 4, stream>>>(qtmp, qkdst);
  }

  // ---- attention: scores + dual softmax (y is dead; M1/M2 overlay it) ----
  k_score_softmax<<<dim3(BH), 256, 256 * 257 * 4, stream>>>(Qbf, Kbf, M1, M2);
  k_vmaps<<<dim3(BH), 256, 2 * 256 * 258 * 2, stream>>>(M1, M2, VL, VR);
  k_xt_blend<<<dim3(BH, 2), 256, 0, stream>>>(xlb, xrb, M1, M2, VL, VR,
                                              x_left, x_right, out);
}
